// EfficientDeformableAttention_17703855194636
// MI455X (gfx1250) — matrix operand order repaired and checked by reading, not compiled, not run on hardware
//
#include <hip/hip_runtime.h>
#include <hip/hip_bf16.h>

// ---------------------------------------------------------------------------
// EfficientDeformableAttention for MI455X (gfx1250, wave32, WMMA)
//
// Pipeline:
//   K1 qproj : query @ [W_off|W_attn] (WMMA bf16)  -> softmax/offset epilogue
//              -> sampInfo[bq][h][p] = {gx, gy, attn_w, 0}
//   K2 vproj : value @ W_val (WMMA bf16) -> vperm[B][NH][HW][HD]  (64MB, L2-resident)
//   K3 sample: wave-per-(q,h), lane-per-channel bilinear gather from vperm
//              -> heads[bq][256]
//   K4 oproj : heads @ W_out (WMMA bf16) -> d_out
// ---------------------------------------------------------------------------

typedef __attribute__((ext_vector_type(16))) __bf16 v16bf;
typedef __attribute__((ext_vector_type(8)))  float  v8f;

#define NHh 8
#define NPp 4
#define HDd 32
#define CC  256
#define HS  128
#define WSs 128
#define HWW (HS * WSs)
#define TK  32

__device__ __forceinline__ v8f wmma_bf16(v16bf a, v16bf b, v8f c) {
  // v_wmma_f32_16x16x32_bf16  (neg_a, A, neg_b, B, c_mod, C, reuse_a, reuse_b)
  return __builtin_amdgcn_wmma_f32_16x16x32_bf16(false, a, false, b, (short)0, c,
                                                 false, false);
}

// A fragment, 16-bit A 16x32 layout (ISA 7.12.2):
//   lane: m = lane&15, half = lane>>4
//   VGPR r (elems 2r,2r+1): K = (r>=4?16:0) + half*8 + (r&3)*2 + pos
// Row-major LDS tile As[m][k] (ld = TK=32, 16B aligned) => two b128 loads.
__device__ __forceinline__ v16bf load_a_frag(const __bf16* row, int half) {
  union { v16bf v; uint4 q[2]; } f;
  f.q[0] = *(const uint4*)(row + half * 8);        // K = half*8 + 0..7
  f.q[1] = *(const uint4*)(row + 16 + half * 8);   // K = 16 + half*8 + 0..7
  return f.v;
}

// B fragment, 32x16 bf16 B (derived from the ISA sparse-B layout):
//   lane: n = lane&15, half = lane>>4; element e: K = (e>=8?16:0) + half*8 + (e&7)
// LDS stores B transposed: Bs[n][k] (ld = TK) => two contiguous b128 loads.
__device__ __forceinline__ v16bf load_b_frag(const __bf16* row, int half) {
  union { v16bf v; uint4 q[2]; } f;
  f.q[0] = *(const uint4*)(row + half * 8);        // K = half*8 + 0..7
  f.q[1] = *(const uint4*)(row + 16 + half * 8);   // K = 16 + half*8 + 0..7
  return f.v;
}

// ---------------------------------------------------------------------------
// K1: q-proj + softmax + sampling-location epilogue.
// Block = 256 thr (8 waves), tile 128 rows x 96 cols, K=256 in steps of 32.
// ---------------------------------------------------------------------------
__global__ __launch_bounds__(256) void qproj_kernel(
    const float* __restrict__ query, const float* __restrict__ refp,
    const float* __restrict__ W_off, const float* __restrict__ b_off,
    const float* __restrict__ W_attn, const float* __restrict__ b_attn,
    float* __restrict__ sampInfo) {
  __shared__ __align__(16) __bf16 As[128 * TK];
  __shared__ __align__(16) __bf16 Bs[96 * TK];
  __shared__ float res[128 * 96];

  const int tid = threadIdx.x;
  const int wave = tid >> 5, lane = tid & 31, half = lane >> 4, l16 = lane & 15;
  const size_t m0 = (size_t)blockIdx.x * 128;

  v8f acc[6] = {};
  for (int k0 = 0; k0 < CC; k0 += TK) {
    for (int i = tid; i < 128 * TK / 4; i += 256) {
      int r = i >> 3, c4 = (i & 7) << 2;
      float4 f = *(const float4*)(query + (m0 + r) * CC + k0 + c4);
      __bf16* d = As + r * TK + c4;
      d[0] = (__bf16)f.x; d[1] = (__bf16)f.y; d[2] = (__bf16)f.z; d[3] = (__bf16)f.w;
    }
    for (int i = tid; i < TK * 96; i += 256) {
      int k = i / 96, n = i % 96;
      float w = (n < 64) ? W_off[(size_t)(k0 + k) * 64 + n]
                         : W_attn[(size_t)(k0 + k) * 32 + (n - 64)];
      Bs[n * TK + k] = (__bf16)w;
    }
    if (k0 + TK < CC)
      __builtin_prefetch(query + (m0 + (tid >> 1)) * CC + k0 + TK + ((tid & 1) << 4), 0, 0);
    __syncthreads();
    v16bf a = load_a_frag(As + (wave * 16 + l16) * TK, half);
#pragma unroll
    for (int nt = 0; nt < 6; ++nt) {
      v16bf bb = load_b_frag(Bs + (nt * 16 + l16) * TK, half);
      acc[nt] = wmma_bf16(a, bb, acc[nt]);
    }
    __syncthreads();
  }

  // Spill accumulators (C/D layout: VGPR r -> m = half*8 + r, n = lane&15).
#pragma unroll
  for (int nt = 0; nt < 6; ++nt) {
    int c = nt * 16 + l16;
#pragma unroll
    for (int r = 0; r < 8; ++r) {
      int m = wave * 16 + half * 8 + r;
      res[m * 96 + c] = acc[nt][r];
    }
  }
  __syncthreads();

  // Epilogue: softmax over NP, offsets -> clamped bilinear pixel coords.
  for (int it = tid; it < 128 * NHh; it += 256) {
    int r = it >> 3, h = it & 7;
    size_t bq = m0 + r;
    float rx = refp[bq * 2 + 0], ry = refp[bq * 2 + 1];
    const float* rr = res + r * 96;
    float l[NPp], wmax = -1e30f;
#pragma unroll
    for (int p = 0; p < NPp; ++p) {
      l[p] = rr[64 + h * NPp + p] + b_attn[h * NPp + p];
      wmax = fmaxf(wmax, l[p]);
    }
    float s = 0.f;
#pragma unroll
    for (int p = 0; p < NPp; ++p) { l[p] = __expf(l[p] - wmax); s += l[p]; }
    float inv = 1.f / s;
    float* out = sampInfo + (bq * NHh + h) * NPp * 4;
#pragma unroll
    for (int p = 0; p < NPp; ++p) {
      int j = (h * NPp + p) * 2;
      float ox = rr[j] + b_off[j];
      float oy = rr[j + 1] + b_off[j + 1];
      float lx = fminf(fmaxf(rx + ox * (0.1f / WSs), 0.f), 1.f);
      float ly = fminf(fmaxf(ry + oy * (0.1f / HS), 0.f), 1.f);
      out[p * 4 + 0] = fminf(fmaxf(lx * WSs - 0.5f, 0.f), WSs - 1.f);
      out[p * 4 + 1] = fminf(fmaxf(ly * HS - 0.5f, 0.f), HS - 1.f);
      out[p * 4 + 2] = l[p] * inv;
      out[p * 4 + 3] = 0.f;
    }
  }
}

// ---------------------------------------------------------------------------
// K2: v-proj GEMM, output permuted to [B, NH, HW, HD].
// Tile 128x64, block = 256 thr (8 waves), wave = 16-row strip x 4 n-subtiles.
// ---------------------------------------------------------------------------
__global__ __launch_bounds__(256) void vproj_kernel(
    const float* __restrict__ value, const float* __restrict__ W_val,
    const float* __restrict__ b_val, float* __restrict__ vperm) {
  __shared__ __align__(16) __bf16 As[128 * TK];
  __shared__ __align__(16) __bf16 Bs[64 * TK];
  const int tid = threadIdx.x;
  const int wave = tid >> 5, lane = tid & 31, half = lane >> 4, l16 = lane & 15;
  const int b = blockIdx.z;
  const size_t m0 = (size_t)blockIdx.x * 128;
  const int n0 = blockIdx.y * 64;
  const float* A = value + (size_t)b * HWW * CC;

  v8f acc[4] = {};
  for (int k0 = 0; k0 < CC; k0 += TK) {
    for (int i = tid; i < 128 * TK / 4; i += 256) {
      int r = i >> 3, c4 = (i & 7) << 2;
      float4 f = *(const float4*)(A + (m0 + r) * CC + k0 + c4);
      __bf16* d = As + r * TK + c4;
      d[0] = (__bf16)f.x; d[1] = (__bf16)f.y; d[2] = (__bf16)f.z; d[3] = (__bf16)f.w;
    }
    for (int i = tid; i < TK * 64; i += 256) {
      int k = i >> 6, n = i & 63;
      Bs[n * TK + k] = (__bf16)W_val[(size_t)(k0 + k) * CC + n0 + n];
    }
    if (k0 + TK < CC)
      __builtin_prefetch(A + (m0 + (tid >> 1)) * CC + k0 + TK + ((tid & 1) << 4), 0, 0);
    __syncthreads();
    v16bf a = load_a_frag(As + (wave * 16 + l16) * TK, half);
#pragma unroll
    for (int nt = 0; nt < 4; ++nt) {
      v16bf bb = load_b_frag(Bs + (nt * 16 + l16) * TK, half);
      acc[nt] = wmma_bf16(a, bb, acc[nt]);
    }
    __syncthreads();
  }

#pragma unroll
  for (int nt = 0; nt < 4; ++nt) {
    int c = n0 + nt * 16 + l16;
    int h = c >> 5, d = c & 31;
    float bias = b_val[c];
    float* ob = vperm + (((size_t)b * NHh + h) * HWW) * HDd + d;
#pragma unroll
    for (int r = 0; r < 8; ++r) {
      size_t m = m0 + wave * 16 + half * 8 + r;
      ob[m * HDd] = acc[nt][r] + bias;
    }
  }
}

// ---------------------------------------------------------------------------
// K3: bilinear gather. Block = one query (8 waves = 8 heads), lane = channel.
// All gathered 128B rows come out of L2 (vperm = 64MB < 192MB L2).
// ---------------------------------------------------------------------------
__global__ __launch_bounds__(256) void sample_kernel(
    const float* __restrict__ vperm, const float* __restrict__ sampInfo,
    float* __restrict__ heads, int Nq) {
  const int bq = blockIdx.x;
  const int b = bq / Nq;
  const int h = threadIdx.x >> 5, d = threadIdx.x & 31;
  const float* base = vperm + (((size_t)b * NHh + h) * HWW) * HDd;
  const float* si = sampInfo + ((size_t)bq * NHh + h) * NPp * 4;
  float acc = 0.f;
#pragma unroll
  for (int p = 0; p < NPp; ++p) {
    float gx = si[p * 4 + 0], gy = si[p * 4 + 1], aw = si[p * 4 + 2];
    float x0f = floorf(gx), y0f = floorf(gy);
    float fx = gx - x0f, fy = gy - y0f;
    int x0 = (int)x0f, y0 = (int)y0f;
    int x1 = x0 + 1 < WSs - 1 ? x0 + 1 : WSs - 1;
    int y1 = y0 + 1 < HS - 1 ? y0 + 1 : HS - 1;
    const float* r00 = base + ((size_t)(y0 * WSs + x0)) * HDd;
    const float* r01 = base + ((size_t)(y0 * WSs + x1)) * HDd;
    const float* r10 = base + ((size_t)(y1 * WSs + x0)) * HDd;
    const float* r11 = base + ((size_t)(y1 * WSs + x1)) * HDd;
    acc += aw * ((1.f - fx) * (1.f - fy) * r00[d] + fx * (1.f - fy) * r01[d] +
                 (1.f - fx) * fy * r10[d] + fx * fy * r11[d]);
  }
  heads[(size_t)bq * CC + h * HDd + d] = acc;
}

// ---------------------------------------------------------------------------
// K4: out-proj GEMM, row-major store into d_out.
// ---------------------------------------------------------------------------
__global__ __launch_bounds__(256) void outproj_kernel(
    const float* __restrict__ Ain, const float* __restrict__ W_out,
    const float* __restrict__ b_out, float* __restrict__ out) {
  __shared__ __align__(16) __bf16 As[128 * TK];
  __shared__ __align__(16) __bf16 Bs[64 * TK];
  const int tid = threadIdx.x;
  const int wave = tid >> 5, lane = tid & 31, half = lane >> 4, l16 = lane & 15;
  const size_t m0 = (size_t)blockIdx.x * 128;
  const int n0 = blockIdx.y * 64;

  v8f acc[4] = {};
  for (int k0 = 0; k0 < CC; k0 += TK) {
    for (int i = tid; i < 128 * TK / 4; i += 256) {
      int r = i >> 3, c4 = (i & 7) << 2;
      float4 f = *(const float4*)(Ain + (m0 + r) * CC + k0 + c4);
      __bf16* d = As + r * TK + c4;
      d[0] = (__bf16)f.x; d[1] = (__bf16)f.y; d[2] = (__bf16)f.z; d[3] = (__bf16)f.w;
    }
    for (int i = tid; i < TK * 64; i += 256) {
      int k = i >> 6, n = i & 63;
      Bs[n * TK + k] = (__bf16)W_out[(size_t)(k0 + k) * CC + n0 + n];
    }
    if (k0 + TK < CC)
      __builtin_prefetch(Ain + (m0 + (tid >> 1)) * CC + k0 + TK + ((tid & 1) << 4), 0, 0);
    __syncthreads();
    v16bf a = load_a_frag(As + (wave * 16 + l16) * TK, half);
#pragma unroll
    for (int nt = 0; nt < 4; ++nt) {
      v16bf bb = load_b_frag(Bs + (nt * 16 + l16) * TK, half);
      acc[nt] = wmma_bf16(a, bb, acc[nt]);
    }
    __syncthreads();
  }

#pragma unroll
  for (int nt = 0; nt < 4; ++nt) {
    int c = n0 + nt * 16 + l16;
    float bias = b_out[c];
#pragma unroll
    for (int r = 0; r < 8; ++r) {
      size_t m = m0 + wave * 16 + half * 8 + r;
      out[m * CC + c] = acc[nt][r] + bias;
    }
  }
}

// ---------------------------------------------------------------------------
extern "C" void kernel_launch(void* const* d_in, const int* in_sizes, int n_in,
                              void* d_out, int out_size, void* d_ws, size_t ws_size,
                              hipStream_t stream) {
  const float* query  = (const float*)d_in[0];
  const float* refp   = (const float*)d_in[1];
  const float* value  = (const float*)d_in[2];
  const float* W_off  = (const float*)d_in[3];
  const float* b_off  = (const float*)d_in[4];
  const float* W_attn = (const float*)d_in[5];
  const float* b_attn = (const float*)d_in[6];
  const float* W_val  = (const float*)d_in[7];
  const float* b_val  = (const float*)d_in[8];
  const float* W_out  = (const float*)d_in[9];
  const float* b_out  = (const float*)d_in[10];
  float* out = (float*)d_out;

  const int BNq = in_sizes[1] / 2;            // B*Nq
  const int B   = in_sizes[2] / (HWW * CC);   // 4
  const int Nq  = BNq / B;

  float* ws       = (float*)d_ws;
  float* sampInfo = ws;                                      // BNq*NH*NP*4 floats
  float* vperm    = sampInfo + (size_t)BNq * NHh * NPp * 4;  // B*NH*HW*HD floats
  float* heads    = vperm + (size_t)B * NHh * HWW * HDd;     // BNq*C floats

  qproj_kernel<<<dim3(BNq / 128), 256, 0, stream>>>(query, refp, W_off, b_off,
                                                    W_attn, b_attn, sampInfo);
  vproj_kernel<<<dim3(HWW / 128, CC / 64, B), 256, 0, stream>>>(value, W_val,
                                                                b_val, vperm);
  sample_kernel<<<dim3(BNq), 256, 0, stream>>>(vperm, sampInfo, heads, Nq);
  outproj_kernel<<<dim3(BNq / 128, CC / 64), 256, 0, stream>>>(heads, W_out,
                                                               b_out, out);
}
